// Local3dAttention_40819369181470
// MI455X (gfx1250) — matrix operand order repaired and checked by reading, not compiled, not run on hardware
//
#include <hip/hip_runtime.h>
#include <hip/hip_bf16.h>

typedef __attribute__((ext_vector_type(16))) _Float16 v16h;
typedef __attribute__((ext_vector_type(8)))  float    v8f;

#define MPOS   16384          // B*S*H*W
#define DIMC   256            // DIM == INNER == 256
#define MTILES (MPOS / 16)    // 1024
#define KCH    8              // 256 / 32 K-chunks
#define NTILES 16             // 256 / 16

// ---------------------------------------------------------------------------
// Swizzle fp32 row-major (M x 256) activation into WMMA A-fragment order:
// Asw[mtile][kc][lane][16 halfs], per CDNA5 ISA 16-bit A 16x32 layout:
//   lane<16 : row m=lane,    h[0..7]=K kb+0..7, h[8..15]=K kb+16..23, kb=0
//   lane>=16: row m=lane-16, kb=8
// One block per mtile; wave w handles K-chunk w.
// ---------------------------------------------------------------------------
__global__ __launch_bounds__(256) void swizzleA_f32(
    const float* __restrict__ A, _Float16* __restrict__ Asw)
{
    const int mtile = blockIdx.x;
    const int kc    = threadIdx.x >> 5;
    const int lane  = threadIdx.x & 31;
    const int m     = mtile * 16 + (lane & 15);
    const int kb    = (lane < 16) ? 0 : 8;
    const float* src = A + (size_t)m * DIMC + kc * 32 + kb;
    v16h hv{};
#pragma unroll
    for (int j = 0; j < 8; ++j) hv[j]     = (_Float16)src[j];
#pragma unroll
    for (int j = 0; j < 8; ++j) hv[8 + j] = (_Float16)src[16 + j];
    *(v16h*)(Asw + ((size_t)(mtile * KCH + kc) * 32 + lane) * 16) = hv;
}

// ---------------------------------------------------------------------------
// Swizzle fp32 weight (256 x 256, [k][n]) into WMMA B-fragment order:
// Bsw[kc][ntile][lane][16 halfs], per ISA B 32x16 layout:
//   lane<16 : col n=lane,    h[0..7]=K 0..7,  h[8..15]=K 16..23
//   lane>=16: col n=lane-16, h[0..7]=K 8..15, h[8..15]=K 24..31
// ---------------------------------------------------------------------------
__global__ __launch_bounds__(256) void swizzleB_f32(
    const float* __restrict__ B, _Float16* __restrict__ Bsw)
{
    const int unit  = blockIdx.x * 8 + (threadIdx.x >> 5);   // 0..127
    const int lane  = threadIdx.x & 31;
    const int kc    = unit >> 4;
    const int ntile = unit & 15;
    const int n     = ntile * 16 + (lane & 15);
    const int krow  = kc * 32 + ((lane < 16) ? 0 : 8);
    v16h hv{};
#pragma unroll
    for (int j = 0; j < 16; ++j)
        hv[j] = (_Float16)B[(size_t)(krow + (j < 8 ? j : 8 + j)) * DIMC + n];
    *(v16h*)(Bsw + ((size_t)(kc * NTILES + ntile) * 32 + lane) * 16) = hv;
}

// ---------------------------------------------------------------------------
// WMMA GEMM: C(M x 256) = Asw(M x 256, f16 frags) * Bsw(256 x 256, f16 frags)
// 256 threads = 8 waves, block tile 64(M) x 256(N); wave tile 32 x 64.
// f32 accumulate in v8f; optional bias; store f16 row-major or f32 row-major.
// ---------------------------------------------------------------------------
__global__ __launch_bounds__(256) void gemm_wmma_f16(
    const _Float16* __restrict__ Asw, const _Float16* __restrict__ Bsw,
    const float* __restrict__ bias,
    _Float16* __restrict__ outH, float* __restrict__ outF)
{
    const int lane  = threadIdx.x & 31;
    const int wid   = threadIdx.x >> 5;
    const int waveM = wid >> 2;                 // 0..1
    const int waveN = wid & 3;                  // 0..3
    const int mt0   = blockIdx.x * 4 + waveM * 2; // first mtile of this wave

    v8f acc[2][4];
#pragma unroll
    for (int i = 0; i < 2; ++i)
#pragma unroll
        for (int j = 0; j < 4; ++j) { v8f z{}; acc[i][j] = z; }

#pragma unroll
    for (int kc = 0; kc < KCH; ++kc) {
        // prefetch next K-chunk of A (global_prefetch_b8)
        if (kc + 1 < KCH)
            __builtin_prefetch(Asw + ((size_t)(mt0 * KCH + kc + 1) * 32 + lane) * 16, 0, 1);

        v16h a[2], b[4];
#pragma unroll
        for (int mt = 0; mt < 2; ++mt)
            a[mt] = *(const v16h*)(Asw + ((size_t)((mt0 + mt) * KCH + kc) * 32 + lane) * 16);
#pragma unroll
        for (int nt = 0; nt < 4; ++nt)
            b[nt] = *(const v16h*)(Bsw + ((size_t)(kc * NTILES + waveN * 4 + nt) * 32 + lane) * 16);

#pragma unroll
        for (int mt = 0; mt < 2; ++mt)
#pragma unroll
            for (int nt = 0; nt < 4; ++nt)
                acc[mt][nt] = __builtin_amdgcn_wmma_f32_16x16x32_f16(
                    false, a[mt], false, b[nt], (short)0, acc[mt][nt], false, false);
    }

    // D layout: VGPR r -> (M = r + (lane<16 ? 0 : 8), N = lane & 15)
    const int rbase = (lane < 16) ? 0 : 8;
    const int coln  = lane & 15;
#pragma unroll
    for (int mt = 0; mt < 2; ++mt) {
        const int m0 = (mt0 + mt) * 16;
#pragma unroll
        for (int nt = 0; nt < 4; ++nt) {
            const int n = (waveN * 4 + nt) * 16 + coln;
            const float bval = bias ? bias[n] : 0.0f;
#pragma unroll
            for (int r = 0; r < 8; ++r) {
                const int m = m0 + rbase + r;
                const float val = acc[mt][nt][r] + bval;
                if (outF) outF[(size_t)m * DIMC + n] = val;
                else      outH[(size_t)m * DIMC + n] = (_Float16)val;
            }
        }
    }
}

// ---------------------------------------------------------------------------
// Local 3D windowed attention core. One thread per (position, head).
// dots over 27 neighbors (fp32 accumulate from f16 loads -> v_fma_mix),
// masked softmax, weighted V sum. Output fp32 row-major (M x 256).
// ---------------------------------------------------------------------------
__global__ __launch_bounds__(256) void attn_local3d(
    const _Float16* __restrict__ qp, const _Float16* __restrict__ kk,
    const _Float16* __restrict__ vv, float* __restrict__ out)
{
    const int gid  = blockIdx.x * 256 + threadIdx.x;
    const int head = gid & 7;
    const int pos  = gid >> 3;
    const int w  = pos & 31;
    const int hh = (pos >> 5) & 31;
    const int s  = (pos >> 10) & 7;
    const int b  = pos >> 13;

    const _Float16* qptr = qp + (size_t)pos * DIMC + head * 32;
    const v16h q0 = *(const v16h*)qptr;
    const v16h q1 = *(const v16h*)(qptr + 16);

    float dots[27];
    int   nposa[27];
    float mx = -1e30f;
#pragma unroll
    for (int idx = 0; idx < 27; ++idx) {
        const int dz = idx / 9 - 1;
        const int dy = (idx / 3) % 3 - 1;
        const int dx = idx % 3 - 1;
        const int zs = s + dz, zh = hh + dy, zw = w + dx;
        const bool valid = ((unsigned)zs < 8u) && ((unsigned)zh < 32u) && ((unsigned)zw < 32u);
        float dot = -1e9f;
        int npos = -1;
        if (valid) {
            npos = ((b * 8 + zs) * 32 + zh) * 32 + zw;
            const _Float16* kp = kk + (size_t)npos * DIMC + head * 32;
            const v16h k0 = *(const v16h*)kp;
            const v16h k1 = *(const v16h*)(kp + 16);
            float a = 0.0f;
#pragma unroll
            for (int j = 0; j < 16; ++j) a += (float)q0[j] * (float)k0[j];
#pragma unroll
            for (int j = 0; j < 16; ++j) a += (float)q1[j] * (float)k1[j];
            dot = a * 0.17677669529663687f;   // 32^-0.5
        }
        dots[idx]  = dot;
        nposa[idx] = npos;
        mx = fmaxf(mx, dot);
    }

    float pbuf[27];
    float sum = 0.0f;
#pragma unroll
    for (int idx = 0; idx < 27; ++idx) {
        const float e = expf(dots[idx] - mx);   // invalid -> exp(-1e9 - mx) == 0
        pbuf[idx] = e;
        sum += e;
    }
    const float inv = 1.0f / sum;

    float acc[32];
#pragma unroll
    for (int j = 0; j < 32; ++j) acc[j] = 0.0f;
#pragma unroll
    for (int idx = 0; idx < 27; ++idx) {
        if (nposa[idx] >= 0) {
            const _Float16* vp = vv + (size_t)nposa[idx] * DIMC + head * 32;
            const v16h v0 = *(const v16h*)vp;
            const v16h v1 = *(const v16h*)(vp + 16);
            const float pw = pbuf[idx] * inv;
#pragma unroll
            for (int j = 0; j < 16; ++j) acc[j]      += pw * (float)v0[j];
#pragma unroll
            for (int j = 0; j < 16; ++j) acc[16 + j] += pw * (float)v1[j];
        }
    }

    float* op = out + (size_t)pos * DIMC + head * 32;
#pragma unroll
    for (int j = 0; j < 32; ++j) op[j] = acc[j];
}

// ---------------------------------------------------------------------------
extern "C" void kernel_launch(void* const* d_in, const int* in_sizes, int n_in,
                              void* d_out, int out_size, void* d_ws, size_t ws_size,
                              hipStream_t stream) {
    const float* x  = (const float*)d_in[0];
    const float* q  = (const float*)d_in[1];
    const float* Wq = (const float*)d_in[2];
    const float* Wk = (const float*)d_in[3];
    const float* Wv = (const float*)d_in[4];
    const float* bv = (const float*)d_in[5];
    const float* Wo = (const float*)d_in[6];
    const float* bo = (const float*)d_in[7];
    float* outp = (float*)d_out;

    const size_t actH = (size_t)MPOS * DIMC * sizeof(_Float16);   // 8 MB
    const size_t wH   = (size_t)DIMC * DIMC * sizeof(_Float16);   // 128 KB
    const size_t actF = (size_t)MPOS * DIMC * sizeof(float);      // 16 MB

    char* p = (char*)d_ws;
    _Float16* AswX = (_Float16*)p; p += actH;
    _Float16* AswQ = (_Float16*)p; p += actH;
    _Float16* AswA = (_Float16*)p; p += actH;
    _Float16* qpH  = (_Float16*)p; p += actH;
    _Float16* kH   = (_Float16*)p; p += actH;
    _Float16* vH   = (_Float16*)p; p += actH;
    _Float16* BswQ = (_Float16*)p; p += wH;
    _Float16* BswK = (_Float16*)p; p += wH;
    _Float16* BswV = (_Float16*)p; p += wH;
    _Float16* BswO = (_Float16*)p; p += wH;
    float*    attnF = (float*)p;   p += actF;

    // 1) fp32 -> f16 WMMA-fragment swizzles
    swizzleA_f32<<<MTILES, 256, 0, stream>>>(x, AswX);
    swizzleA_f32<<<MTILES, 256, 0, stream>>>(q, AswQ);
    swizzleB_f32<<<16, 256, 0, stream>>>(Wq, BswQ);
    swizzleB_f32<<<16, 256, 0, stream>>>(Wk, BswK);
    swizzleB_f32<<<16, 256, 0, stream>>>(Wv, BswV);
    swizzleB_f32<<<16, 256, 0, stream>>>(Wo, BswO);

    // 2) projections via v_wmma_f32_16x16x32_f16 (f32 accumulate)
    gemm_wmma_f16<<<MPOS / 64, 256, 0, stream>>>(AswQ, BswQ, nullptr, qpH, nullptr);
    gemm_wmma_f16<<<MPOS / 64, 256, 0, stream>>>(AswX, BswK, nullptr, kH, nullptr);
    gemm_wmma_f16<<<MPOS / 64, 256, 0, stream>>>(AswX, BswV, bv, vH, nullptr);

    // 3) local windowed attention (fp32 softmax)
    attn_local3d<<<(MPOS * 8) / 256, 256, 0, stream>>>(qpH, kH, vH, attnF);

    // 4) output projection: attn @ Wo + bo -> d_out (fp32)
    swizzleA_f32<<<MTILES, 256, 0, stream>>>(attnF, AswA);
    gemm_wmma_f16<<<MPOS / 64, 256, 0, stream>>>(AswA, BswO, bo, nullptr, outp);
}
